// SegmentEncoder_48198122996212
// MI455X (gfx1250) — matrix operand order repaired and checked by reading, not compiled, not run on hardware
//
#include <hip/hip_runtime.h>
#include <hip/hip_bf16.h>
#include <math.h>

#define HIDDEN 128

typedef __attribute__((ext_vector_type(16))) __bf16 v16bf;
typedef __attribute__((ext_vector_type(8)))  __bf16 v8bf;
typedef __attribute__((ext_vector_type(8)))  float  v8f;

// f32 atomic add straight to L2 (out array fits in MI455X's 192MB L2)
__device__ __forceinline__ void gatomic_add_f32(float* p, float v) {
    asm volatile("global_atomic_add_f32 %0, %1, off" :: "v"(p), "v"(v) : "memory");
}

__device__ __forceinline__ float gelu_exact(float y) {
    return 0.5f * y * (1.0f + erff(y * 0.70710678118654752f));
}

// ---------------- degree / normalization ----------------
__global__ void k_deg_init(float* deg, int n) {
    int i = blockIdx.x * blockDim.x + threadIdx.x;
    if (i < n) deg[i] = 1.0f;   // self-loop contributes 1
}

__global__ void k_deg_edges(const long long* __restrict__ dst, float* deg, int E) {
    int e = blockIdx.x * blockDim.x + threadIdx.x;
    if (e < E) gatomic_add_f32(&deg[(int)dst[e]], 1.0f);
}

__global__ void k_deg_inv(float* deg, int n) {
    int i = blockIdx.x * blockDim.x + threadIdx.x;
    if (i < n) { float d = deg[i]; deg[i] = (d > 0.0f) ? rsqrtf(d) : 0.0f; }
}

// ---------------- bf16 hi/lo split of activations ----------------
__global__ void k_split_bf16(const float* __restrict__ x, __bf16* __restrict__ hi,
                             __bf16* __restrict__ lo, int n) {
    int i = blockIdx.x * blockDim.x + threadIdx.x;
    if (i < n) {
        float a = x[i];
        __bf16 h = (__bf16)a;
        hi[i] = h;
        lo[i] = (__bf16)(a - (float)h);
    }
}

// Pack W[128x128] into the CDNA5 B-matrix (32x16 bf16, K striped across lanes
// within a VGPR) lane layout so GEMM waves load contiguous 32B vectors.
// idx = ((nt*4 + kt)*32 + lane)*16 + e ; element e of lane -> B[K=8*(lane>>4)+(e&7)+16*(e>>3), N]
__global__ void k_pack_w(const float* __restrict__ W, __bf16* __restrict__ bhi,
                         __bf16* __restrict__ blo) {
    int idx = blockIdx.x * blockDim.x + threadIdx.x;
    if (idx >= 8 * 4 * 32 * 16) return;
    int e    = idx & 15;
    int lane = (idx >> 4) & 31;
    int kt   = (idx >> 9) & 3;
    int nt   = idx >> 11;
    int hf   = lane >> 4;
    int n    = nt * 16 + (lane & 15);
    int k    = kt * 32 + 8 * hf + (e & 7) + 16 * (e >> 3);
    float a = W[k * HIDDEN + n];
    __bf16 h = (__bf16)a;
    bhi[idx] = h;
    blo[idx] = (__bf16)(a - (float)h);
}

// ---------------- WMMA GEMM: C[M x 128] = A[M x 128] * W[128 x 128] ----------------
// One workgroup (8 waves) per 16-row M tile; wave w owns N-tile w.
// K=128 -> 4 steps of v_wmma_f32_16x16x32_bf16, with hi/lo error correction
// (Ahi*Bhi + Ahi*Blo + Alo*Bhi) for near-f32 accuracy.
__global__ void __launch_bounds__(256) k_gemm_wmma(
        const __bf16* __restrict__ Ahi, const __bf16* __restrict__ Alo,
        const __bf16* __restrict__ Bhi, const __bf16* __restrict__ Blo,
        float* __restrict__ C, int M) {
    const int wave = threadIdx.x >> 5;        // N tile 0..7
    const int lane = threadIdx.x & 31;
    const int hf   = lane >> 4;               // lane half selects K sub-bands
    const int mr   = lane & 15;

    long row = (long)blockIdx.x * 16 + mr;    // A layout: M = lane&15 for both halves
    long rowc = row < (long)M ? row : (long)M - 1;   // clamp loads, keep EXEC all-1s

    v8f acc = {};
    #pragma unroll
    for (int kt = 0; kt < 4; ++kt) {
        // A 16x32 bf16 tile: lane covers K = {8h..8h+7, 8h+16..8h+23} -> two 16B runs
        const long abase = rowc * HIDDEN + kt * 32 + 8 * hf;
        v8bf a0h = *(const v8bf*)(Ahi + abase);
        v8bf a1h = *(const v8bf*)(Ahi + abase + 16);
        v8bf a0l = *(const v8bf*)(Alo + abase);
        v8bf a1l = *(const v8bf*)(Alo + abase + 16);
        v16bf ah, al;
        #pragma unroll
        for (int i = 0; i < 8; ++i) {
            ah[i] = a0h[i]; ah[8 + i] = a1h[i];
            al[i] = a0l[i]; al[8 + i] = a1l[i];
        }
        // B pre-packed: one contiguous 32B vector per lane per K step
        const int bidx = ((wave * 4 + kt) * 32 + lane) * 16;
        v16bf bh = *(const v16bf*)(Bhi + bidx);
        v16bf bl = *(const v16bf*)(Blo + bidx);

        acc = __builtin_amdgcn_wmma_f32_16x16x32_bf16(false, ah, false, bh, (short)0, acc, false, false);
        acc = __builtin_amdgcn_wmma_f32_16x16x32_bf16(false, ah, false, bl, (short)0, acc, false, false);
        acc = __builtin_amdgcn_wmma_f32_16x16x32_bf16(false, al, false, bh, (short)0, acc, false, false);
    }

    // D layout: VGPR r, lane -> C[m0 + r + 8*(lane>>4)][n0 + (lane&15)]
    const int n = wave * 16 + mr;
    #pragma unroll
    for (int r = 0; r < 8; ++r) {
        long m = (long)blockIdx.x * 16 + r + 8 * hf;
        if (m < (long)M) C[m * HIDDEN + n] = acc[r];
    }
}

// ---------------- self-loop term + bias (also initializes the accumulator) ----------------
__global__ void k_selfloop_bias(const float* __restrict__ hmat, const float* __restrict__ dinv,
                                const float* __restrict__ b, float* __restrict__ out, int n) {
    int idx = blockIdx.x * blockDim.x + threadIdx.x;
    if (idx < n * HIDDEN) {
        int i = idx >> 7, f = idx & 127;
        float di = dinv[i];
        out[idx] = hmat[idx] * di * di + b[f];
    }
}

// ---------------- edge gather + normalized scatter-add (the bandwidth hot spot) ----------------
// One wave32 per edge: each lane moves a float4 slice of the 512B feature row.
__global__ void k_edge_agg(const long long* __restrict__ src, const long long* __restrict__ dst,
                           const float* __restrict__ dinv, const float* __restrict__ hmat,
                           float* __restrict__ out, int E) {
    int gid = blockIdx.x * blockDim.x + threadIdx.x;
    int e    = gid >> 5;
    int subl = gid & 31;
    if (e >= E) return;
    int s = (int)src[e], d = (int)dst[e];
    float w = dinv[s] * dinv[d];
    const float4 v = *(const float4*)(hmat + (long)s * HIDDEN + subl * 4);
    float* p = out + (long)d * HIDDEN + subl * 4;
    gatomic_add_f32(p + 0, v.x * w);
    gatomic_add_f32(p + 1, v.y * w);
    gatomic_add_f32(p + 2, v.z * w);
    gatomic_add_f32(p + 3, v.w * w);
}

// ---------------- LayerNorm + exact GELU, one wave32 per node ----------------
__global__ void k_ln_gelu(const float* __restrict__ in, const float* __restrict__ g,
                          const float* __restrict__ be, float* __restrict__ out, int n) {
    int gid  = blockIdx.x * blockDim.x + threadIdx.x;
    int node = gid >> 5;
    int lane = gid & 31;
    if (node >= n) return;
    const float4 v = *(const float4*)(in + (long)node * HIDDEN + lane * 4);
    float s  = v.x + v.y + v.z + v.w;
    float sq = v.x * v.x + v.y * v.y + v.z * v.z + v.w * v.w;
    #pragma unroll
    for (int off = 16; off; off >>= 1) {
        s  += __shfl_xor(s,  off, 32);
        sq += __shfl_xor(sq, off, 32);
    }
    float mu  = s * (1.0f / 128.0f);
    float var = sq * (1.0f / 128.0f) - mu * mu;
    float rs  = rsqrtf(var + 1e-5f);
    int f = lane * 4;
    float4 o;
    o.x = gelu_exact((v.x - mu) * rs * g[f + 0] + be[f + 0]);
    o.y = gelu_exact((v.y - mu) * rs * g[f + 1] + be[f + 1]);
    o.z = gelu_exact((v.z - mu) * rs * g[f + 2] + be[f + 2]);
    o.w = gelu_exact((v.w - mu) * rs * g[f + 3] + be[f + 3]);
    *(float4*)(out + (long)node * HIDDEN + f) = o;
}

extern "C" void kernel_launch(void* const* d_in, const int* in_sizes, int n_in,
                              void* d_out, int out_size, void* d_ws, size_t ws_size,
                              hipStream_t stream) {
    const float*     x   = (const float*)d_in[0];
    const long long* ei  = (const long long*)d_in[1];
    const float*     W1  = (const float*)d_in[2];
    const float*     b1  = (const float*)d_in[3];
    const float*     W2  = (const float*)d_in[4];
    const float*     b2  = (const float*)d_in[5];
    const float*     g1  = (const float*)d_in[6];
    const float*     be1 = (const float*)d_in[7];
    const float*     g2  = (const float*)d_in[8];
    const float*     be2 = (const float*)d_in[9];
    float* out = (float*)d_out;

    const int N  = in_sizes[0] / HIDDEN;
    const int E  = in_sizes[1] / 2;
    const long long* src = ei;
    const long long* dst = ei + E;
    const int NH = N * HIDDEN;
    const int MT = (N + 15) / 16;

    char* ws = (char*)d_ws;
    size_t off = 0;
    auto walloc = [&](size_t bytes) -> void* {
        void* p = ws + off;
        off = (off + bytes + 255) & ~(size_t)255;
        return p;
    };
    float*  buf1 = (float*) walloc((size_t)NH * sizeof(float));   // h = A*W
    float*  buf2 = (float*) walloc((size_t)NH * sizeof(float));   // aggregate / activations
    float*  dinv = (float*) walloc((size_t)N * sizeof(float));
    __bf16* Ahi  = (__bf16*)walloc((size_t)NH * sizeof(__bf16));
    __bf16* Alo  = (__bf16*)walloc((size_t)NH * sizeof(__bf16));
    __bf16* Bhi  = (__bf16*)walloc((size_t)HIDDEN * HIDDEN * sizeof(__bf16));
    __bf16* Blo  = (__bf16*)walloc((size_t)HIDDEN * HIDDEN * sizeof(__bf16));

    const int T = 256;
    const int gN   = (N + T - 1) / T;
    const int gE   = (E + T - 1) / T;
    const int gNH  = (NH + T - 1) / T;
    const int gW   = (HIDDEN * HIDDEN + T - 1) / T;
    const int gE32 = (int)(((long long)E * 32 + T - 1) / T);
    const int gN32 = (int)(((long long)N * 32 + T - 1) / T);

    // degrees (shared by both layers)
    k_deg_init <<<gN, T, 0, stream>>>(dinv, N);
    k_deg_edges<<<gE, T, 0, stream>>>(dst, dinv, E);
    k_deg_inv  <<<gN, T, 0, stream>>>(dinv, N);

    // ---- layer 1 ----
    k_split_bf16<<<gNH, T, 0, stream>>>(x, Ahi, Alo, NH);
    k_pack_w    <<<gW,  T, 0, stream>>>(W1, Bhi, Blo);
    k_gemm_wmma <<<MT, 256, 0, stream>>>(Ahi, Alo, Bhi, Blo, buf1, N);
    k_selfloop_bias<<<gNH, T, 0, stream>>>(buf1, dinv, b1, buf2, N);
    k_edge_agg  <<<gE32, T, 0, stream>>>(src, dst, dinv, buf1, buf2, E);
    k_ln_gelu   <<<gN32, T, 0, stream>>>(buf2, g1, be1, buf2, N);   // in-place ok (wave-private rows)

    // ---- layer 2 ----
    k_split_bf16<<<gNH, T, 0, stream>>>(buf2, Ahi, Alo, NH);
    k_pack_w    <<<gW,  T, 0, stream>>>(W2, Bhi, Blo);
    k_gemm_wmma <<<MT, 256, 0, stream>>>(Ahi, Alo, Bhi, Blo, buf1, N);
    k_selfloop_bias<<<gNH, T, 0, stream>>>(buf1, dinv, b2, buf2, N);
    k_edge_agg  <<<gE32, T, 0, stream>>>(src, dst, dinv, buf1, buf2, E);
    k_ln_gelu   <<<gN32, T, 0, stream>>>(buf2, g2, be2, out, N);
}
